// SAG_atten_up_14293651161269
// MI455X (gfx1250) — matrix operand order repaired and checked by reading, not compiled, not run on hardware
//
#include <hip/hip_runtime.h>
#include <math.h>

typedef __attribute__((ext_vector_type(16))) _Float16 v16h;
typedef __attribute__((ext_vector_type(8)))  _Float16 v8h;
typedef __attribute__((ext_vector_type(8)))  float    v8f;
typedef __attribute__((ext_vector_type(4)))  int      v4i;

// Problem dims (fixed by the reference)
#define BB   4
#define CC   128
#define HH   256
#define WW   256
#define HWsz 65536          // HH*WW
#define NPIX (BB * HWsz)    // 262144

__device__ __forceinline__ float silu_f(float v) { return v / (1.0f + __expf(-v)); }
__device__ __forceinline__ float gelu_f(float v) { return 0.5f * v * (1.0f + erff(v * 0.70710678118654752f)); }

// ---- Optional gfx1250 async global->LDS copy (guarded; falls back) -------
#if defined(__has_builtin)
#  if __has_builtin(__builtin_amdgcn_global_load_async_to_lds_b128)
#    define USE_ASYNC_LDS 1
#  endif
#endif
#ifndef USE_ASYNC_LDS
#  define USE_ASYNC_LDS 0
#endif

typedef __attribute__((address_space(1))) v4i* gv4i_p;  // global int4*
typedef __attribute__((address_space(3))) v4i* lv4i_p;  // LDS int4*

__device__ __forceinline__ void copy16_to_lds(const _Float16* g, _Float16* l) {
#if USE_ASYNC_LDS
    __builtin_amdgcn_global_load_async_to_lds_b128(
        (gv4i_p)(v4i*)g, (lv4i_p)(v4i*)l, 0, 0);
#else
    *(v16h*)l = *(const v16h*)g;
#endif
}
__device__ __forceinline__ void copy_to_lds_wait() {
#if USE_ASYNC_LDS
#  if __has_builtin(__builtin_amdgcn_s_wait_asynccnt)
    __builtin_amdgcn_s_wait_asynccnt(0);
#  else
    asm volatile("s_wait_asynccnt 0x0" ::: "memory");
#  endif
#endif
}

// ---- WMMA fragment loaders (contiguous k = 16h + s order in A and B, wave32) -------------------
// A (16x32 f16): lanes 0-15 hold K{0..15} of row M=lane; lanes 16-31 hold K{16..31}.
__device__ __forceinline__ v16h load_a_frag(const _Float16* A, int row, int ldk) {
    int lane = threadIdx.x & 31;
    const _Float16* p = A + (size_t)(row + (lane & 15)) * ldk + ((lane >> 4) << 4);
    v8h lo = *(const v8h*)p;          // K k0..k0+7
    v8h hi = *(const v8h*)(p + 8);    // K k0+8..k0+15
    v16h r;
#pragma unroll
    for (int i = 0; i < 8; ++i) { r[i] = lo[i]; r[i + 8] = hi[i]; }
    return r;
}
// B (32x16 f16): lane holds column n=lane&15, K base 0 (lanes 0-15) or 16 (lanes 16-31).
// Requires tile stored n-major: Bt[n][k], so one lane reads 16 contiguous f16 (32B).
__device__ __forceinline__ v16h load_b_frag(const _Float16* Bt, int col0, int ldk) {
    int lane = threadIdx.x & 31;
    return *(const v16h*)(Bt + (size_t)(col0 + (lane & 15)) * ldk + ((lane >> 4) << 4));
}

// ---- Stage 0: block-norm statistics --------------------------------------
__global__ void k_zero_sums(float* sums) {
    if (threadIdx.x < 21) sums[threadIdx.x] = 0.0f;
}

__global__ void k_block_sums(const float* __restrict__ x, float* __restrict__ sums, int n) {
    __shared__ float ls[21];
    if (threadIdx.x < 21) ls[threadIdx.x] = 0.0f;
    __syncthreads();
    int stride = gridDim.x * blockDim.x;
    for (int i = blockIdx.x * blockDim.x + threadIdx.x; i < n; i += stride) {
        float v = x[i];
        v = v > 0.0f ? v : 0.0f;
        int p  = i & (HWsz - 1);
        int yy = p >> 8, xx = p & 255;
        atomicAdd(&ls[0], v);
        atomicAdd(&ls[1 + ((yy >> 7) << 1) + (xx >> 7)], v);
        atomicAdd(&ls[5 + ((yy >> 6) << 2) + (xx >> 6)], v);
    }
    __syncthreads();
    if (threadIdx.x < 21) atomicAdd(&sums[threadIdx.x], ls[threadIdx.x]);
}

__global__ void k_finalize_inv(const float* __restrict__ sums, float* __restrict__ invv) {
    int t = threadIdx.x;
    if (t >= 21) return;
    float cnt = (t == 0) ? 33554432.0f : (t < 5 ? 8388608.0f : 2097152.0f);
    invv[t] = 1.0f / (sums[t] / cnt + 1e-4f);
}

// ---- Weight conversion (f32 -> f16, mix conv reordered tap-major) --------
#define WOFF1 49152                    // 384*128  (wcat)
#define WOFF2 (WOFF1 + 589824)         // 128*4608 (wm16)
#define WOFF3 (WOFF2 + 16384)          // 128*128  (wa16)
#define WTOT  (WOFF3 + 16384)          // 128*128  (wb16)
__global__ void k_cvt_weights(const float* __restrict__ w1, const float* __restrict__ w2,
                              const float* __restrict__ w3, const float* __restrict__ wm,
                              const float* __restrict__ wa, const float* __restrict__ wb,
                              _Float16* __restrict__ wcat, _Float16* __restrict__ wm16,
                              _Float16* __restrict__ wa16, _Float16* __restrict__ wb16) {
    int i = blockIdx.x * blockDim.x + threadIdx.x;
    if (i >= WTOT) return;
    if (i < WOFF1) {
        int row = i >> 7, col = i & 127;
        const float* w = (row < 128) ? w1 : ((row < 256) ? w2 : w3);
        wcat[i] = (_Float16)w[(row & 127) * 128 + col];
    } else if (i < WOFF2) {
        int j = i - WOFF1;
        int co = j / 4608, r = j % 4608;
        int tap = r >> 9, ci = r & 511;                 // k = tap*512 + ci
        wm16[j] = (_Float16)wm[co * 4608 + ci * 9 + tap];
    } else if (i < WOFF3) {
        int j = i - WOFF2;
        wa16[j] = (_Float16)wa[j];
    } else {
        int j = i - WOFF3;
        wb16[j] = (_Float16)wb[j];
    }
}

// ---- Pack residual x (f16) into channels 0..127 of the concat buffer -----
__global__ void k_pack_x(const float* __restrict__ x, _Float16* __restrict__ y4c, int n) {
    int i = blockIdx.x * blockDim.x + threadIdx.x;
    if (i >= n) return;
    int b = i >> 23;                  // 128*HW = 2^23
    int rem = i & ((1 << 23) - 1);
    y4c[((size_t)b << 25) + rem] = (_Float16)x[i];   // 512*HW = 2^25
}

// ---- Stage 1: fused s1/s2/s4 1x1 convs (stacked 384x128 GEMM) ------------
// grid = (NPIX/64, 3); block = 256 (8 waves). Wave w -> M rows 16w..16w+15, N=64.
__global__ void __launch_bounds__(256)
k_scale_convs(const float* __restrict__ x, const _Float16* __restrict__ wcat,
              const float* __restrict__ g1, const float* __restrict__ b1,
              const float* __restrict__ g2, const float* __restrict__ b2,
              const float* __restrict__ g3, const float* __restrict__ b3,
              const float* __restrict__ invv, _Float16* __restrict__ y4c) {
    __shared__ __align__(32) _Float16 As[128][32];
    __shared__ __align__(32) _Float16 Bt[64][32];   // n-major
    __shared__ float sG[128], sS[128];

    int sidx = blockIdx.y;            // 0:s1 1:s2 2:s4
    int m0   = sidx << 7;             // row offset into wcat (0/128/256)
    int p0   = blockIdx.x << 6;       // 64-pixel tile (never straddles batch)
    int b    = p0 >> 16;
    int pin  = p0 & (HWsz - 1);
    int t    = threadIdx.x, w = t >> 5, lane = t & 31;

    const float* gs = (sidx == 0) ? g1 : ((sidx == 1) ? g2 : g3);
    const float* ss = (sidx == 0) ? b1 : ((sidx == 1) ? b2 : b3);
    if (t < 128) { sG[t] = gs[t]; sS[t] = ss[t]; }

    v8f acc[4];
#pragma unroll
    for (int s = 0; s < 4; ++s)
#pragma unroll
        for (int i = 0; i < 8; ++i) acc[s][i] = 0.0f;

    for (int kc = 0; kc < 4; ++kc) {               // K = 128 in chunks of 32
        __syncthreads();
        {   // A: 128x32 f16 from stacked weights
            int mm = t >> 1, k16 = (t & 1) << 4;
            *(v16h*)&As[mm][k16] =
                *(const v16h*)(wcat + (size_t)(m0 + mm) * 128 + kc * 32 + k16);
        }
        {   // B: relu(x), f32->f16 on the fly, stored n-major
            int kk = t >> 3, n0 = (t & 7) << 3;
            const float* src = x + (size_t)b * CC * HWsz + (size_t)(kc * 32 + kk) * HWsz + pin + n0;
#pragma unroll
            for (int j = 0; j < 8; ++j) {
                float v = src[j];
                Bt[n0 + j][kk] = (_Float16)(v > 0.0f ? v : 0.0f);
            }
        }
        __syncthreads();
        v16h a = load_a_frag(&As[0][0], 16 * w, 32);
#pragma unroll
        for (int s = 0; s < 4; ++s) {
            v16h bf = load_b_frag(&Bt[0][0], 16 * s, 32);
            acc[s] = __builtin_amdgcn_wmma_f32_16x16x32_f16(
                false, a, false, bf, (short)0, acc[s], false, false);
        }
    }
    __syncthreads();

    // Epilogue: fold 1/(blockmean+eps), gamma/beta, SiLU; write f16 concat buffer.
#pragma unroll
    for (int s = 0; s < 4; ++s) {
        int n  = 16 * s + (lane & 15);
        int p  = pin + n;
        int yy = p >> 8, xx = p & 255;
        float iv;
        if (sidx == 0)      iv = invv[0];
        else if (sidx == 1) iv = invv[1 + ((yy >> 7) << 1) + (xx >> 7)];
        else                iv = invv[5 + ((yy >> 6) << 2) + (xx >> 6)];
#pragma unroll
        for (int i = 0; i < 8; ++i) {
            int c = 16 * w + i + ((lane >> 4) << 3);
            float v = acc[s][i] * iv * sG[c] + sS[c];
            v = silu_f(v);
            y4c[(size_t)b * 512 * HWsz + (size_t)(128 + m0 + c) * HWsz + p] = (_Float16)v;
        }
    }
}

// ---- Stage 2: mix 3x3 conv, 4C->C (implicit im2col GEMM, K=4608) ---------
// N-tile = 128 pixels (one half-row). For each (dy, 32ch chunk) the halo
// window (32ch x 130px) is staged ONCE and reused by all three dx taps via a
// shifted column offset in the B-fragment load. All three taps' A chunks are
// staged together (async global->LDS when available).
// grid = B*H*(W/128) = 2048; block = 256 (8 waves); acc = 8 subtiles/wave.
__global__ void __launch_bounds__(256)
k_mix(const _Float16* __restrict__ y4c, const _Float16* __restrict__ wm16,
      const float* __restrict__ gm, const float* __restrict__ bm,
      _Float16* __restrict__ z16) {
    __shared__ __align__(32) _Float16 As3[3][128][32];  // taps dx=-1,0,1
    __shared__ __align__(32) _Float16 Bt[136][32];      // n-major halo window
    __shared__ float sG[128], sS[128];

    int tile = blockIdx.x;
    int x0 = (tile & 1) << 7;         // 0 or 128
    int y0 = (tile >> 1) & 255;
    int b  = tile >> 9;
    int t  = threadIdx.x, w = t >> 5, lane = t & 31;
    if (t < 128) { sG[t] = gm[t]; sS[t] = bm[t]; }

    size_t bOff = (size_t)b * 512 * HWsz;
    v8f acc[8];
#pragma unroll
    for (int s = 0; s < 8; ++s)
#pragma unroll
        for (int i = 0; i < 8; ++i) acc[s][i] = 0.0f;

    for (int dyi = 0; dyi < 3; ++dyi) {
        int yr = y0 + dyi - 1;
        bool rowok = (yr >= 0) && (yr < HH);
        for (int cc = 0; cc < 16; ++cc) {
            int ci0 = cc << 5;
            __syncthreads();
            {   // B halo window: pixels x0-1 .. x0+128 (130 valid, pad to 136)
                int kk = t >> 3, g = t & 7;
                const _Float16* src = y4c + bOff + (size_t)(ci0 + kk) * HWsz + (size_t)(rowok ? yr : 0) * WW;
#pragma unroll
                for (int j = 0; j < 17; ++j) {
                    int n  = g * 17 + j;
                    int xx = x0 - 1 + n;
                    _Float16 v = (_Float16)0.0f;
                    if (rowok && n < 130 && xx >= 0 && xx < WW) v = src[xx];
                    Bt[n][kk] = v;
                }
            }
            {   // A chunks for the 3 taps of this dy (pure 16B copies)
                int mm = t >> 1, k16 = (t & 1) << 4;
#pragma unroll
                for (int dxi = 0; dxi < 3; ++dxi) {
                    int tap = dyi * 3 + dxi;
                    copy16_to_lds(wm16 + (size_t)mm * 4608 + tap * 512 + ci0 + k16,
                                  &As3[dxi][mm][k16]);
                }
            }
            copy_to_lds_wait();
            __syncthreads();
#pragma unroll
            for (int dxi = 0; dxi < 3; ++dxi) {
                v16h a = load_a_frag(&As3[dxi][0][0], 16 * w, 32);
#pragma unroll
                for (int s = 0; s < 8; ++s) {
                    // output pixel np = 16s + lane%16 needs window row np+dxi
                    v16h bf = load_b_frag(&Bt[0][0], dxi + 16 * s, 32);
                    acc[s] = __builtin_amdgcn_wmma_f32_16x16x32_f16(
                        false, a, false, bf, (short)0, acc[s], false, false);
                }
            }
        }
    }
    __syncthreads();

#pragma unroll
    for (int s = 0; s < 8; ++s) {
        int p = y0 * WW + x0 + 16 * s + (lane & 15);
#pragma unroll
        for (int i = 0; i < 8; ++i) {
            int c = 16 * w + i + ((lane >> 4) << 3);
            float v = acc[s][i] * sG[c] + sS[c];
            v = silu_f(v);
            z16[(size_t)b * CC * HWsz + (size_t)c * HWsz + p] = (_Float16)v;
        }
    }
}

// ---- Stage 3: fused reweight MLP + residual multiply ---------------------
// GEMM(wa) -> SiLU -> GELU -> LDS transpose -> GEMM(wb) -> SiLU -> *x -> out
__global__ void __launch_bounds__(256)
k_mlp_out(const _Float16* __restrict__ z16, const _Float16* __restrict__ wa16,
          const _Float16* __restrict__ wb16,
          const float* __restrict__ ga, const float* __restrict__ ba,
          const float* __restrict__ gb, const float* __restrict__ bb,
          const float* __restrict__ x, float* __restrict__ out) {
    __shared__ __align__(32) _Float16 Bt[64][32];
    __shared__ __align__(32) _Float16 Ts[64][128];   // intermediate, n-major
    __shared__ float sGa[128], sBa[128], sGb[128], sBb[128];

    int p0  = blockIdx.x << 6;
    int b   = p0 >> 16;
    int pin = p0 & (HWsz - 1);
    int t   = threadIdx.x, w = t >> 5, lane = t & 31;
    if (t < 128) { sGa[t] = ga[t]; sBa[t] = ba[t]; sGb[t] = gb[t]; sBb[t] = bb[t]; }

    v8f acc[4];
#pragma unroll
    for (int s = 0; s < 4; ++s)
#pragma unroll
        for (int i = 0; i < 8; ++i) acc[s][i] = 0.0f;

    // GEMM a: A frags straight from global (K=128, 4 chunks)
    for (int kc = 0; kc < 4; ++kc) {
        __syncthreads();
        {
            int kk = t >> 3, n0 = (t & 7) << 3;
            const _Float16* src = z16 + (size_t)b * CC * HWsz + (size_t)(kc * 32 + kk) * HWsz + pin + n0;
            v8h v = *(const v8h*)src;
#pragma unroll
            for (int j = 0; j < 8; ++j) Bt[n0 + j][kk] = v[j];
        }
        __syncthreads();
        v16h a = load_a_frag(wa16 + kc * 32, 16 * w, 128);
#pragma unroll
        for (int s = 0; s < 4; ++s) {
            v16h bf = load_b_frag(&Bt[0][0], 16 * s, 32);
            acc[s] = __builtin_amdgcn_wmma_f32_16x16x32_f16(
                false, a, false, bf, (short)0, acc[s], false, false);
        }
    }
    __syncthreads();

    // SiLU -> GELU, transpose-store to Ts[n][c]
#pragma unroll
    for (int s = 0; s < 4; ++s) {
        int n = 16 * s + (lane & 15);
#pragma unroll
        for (int i = 0; i < 8; ++i) {
            int c = 16 * w + i + ((lane >> 4) << 3);
            float v = acc[s][i] * sGa[c] + sBa[c];
            v = gelu_f(silu_f(v));
            Ts[n][c] = (_Float16)v;
        }
    }
    __syncthreads();

#pragma unroll
    for (int s = 0; s < 4; ++s)
#pragma unroll
        for (int i = 0; i < 8; ++i) acc[s][i] = 0.0f;

    // GEMM b: B frags from Ts (already n-major, ldk=128)
    for (int kc = 0; kc < 4; ++kc) {
        v16h a = load_a_frag(wb16 + kc * 32, 16 * w, 128);
#pragma unroll
        for (int s = 0; s < 4; ++s) {
            v16h bf = load_b_frag(&Ts[0][0] + kc * 32, 16 * s, 128);
            acc[s] = __builtin_amdgcn_wmma_f32_16x16x32_f16(
                false, a, false, bf, (short)0, acc[s], false, false);
        }
    }

    // SiLU, multiply residual, f32 out
#pragma unroll
    for (int s = 0; s < 4; ++s) {
        int p = pin + 16 * s + (lane & 15);
#pragma unroll
        for (int i = 0; i < 8; ++i) {
            int c = 16 * w + i + ((lane >> 4) << 3);
            float v = acc[s][i] * sGb[c] + sBb[c];
            v = silu_f(v);
            size_t gi = (size_t)b * CC * HWsz + (size_t)c * HWsz + p;
            out[gi] = v * x[gi];
        }
    }
}

// ---- Host-side orchestration ---------------------------------------------
extern "C" void kernel_launch(void* const* d_in, const int* in_sizes, int n_in,
                              void* d_out, int out_size, void* d_ws, size_t ws_size,
                              hipStream_t stream) {
    (void)in_sizes; (void)n_in; (void)out_size; (void)ws_size;
    const float* x  = (const float*)d_in[0];
    const float* w1 = (const float*)d_in[1];
    const float* g1 = (const float*)d_in[2];
    const float* b1 = (const float*)d_in[3];
    const float* w2 = (const float*)d_in[4];
    const float* g2 = (const float*)d_in[5];
    const float* b2 = (const float*)d_in[6];
    const float* w3 = (const float*)d_in[7];
    const float* g3 = (const float*)d_in[8];
    const float* b3 = (const float*)d_in[9];
    const float* wm = (const float*)d_in[10];
    const float* gm = (const float*)d_in[11];
    const float* bm = (const float*)d_in[12];
    const float* wa = (const float*)d_in[13];
    const float* ga = (const float*)d_in[14];
    const float* ba = (const float*)d_in[15];
    const float* wb = (const float*)d_in[16];
    const float* gb = (const float*)d_in[17];
    const float* bb = (const float*)d_in[18];
    float* out = (float*)d_out;

    char* ws = (char*)d_ws;
    size_t off = 0;
    _Float16* y4c  = (_Float16*)(ws + off); off += (size_t)BB * 512 * HWsz * 2; // concat (f16)
    _Float16* z16  = (_Float16*)(ws + off); off += (size_t)BB * CC  * HWsz * 2; // mix out (f16)
    _Float16* wm16 = (_Float16*)(ws + off); off += (size_t)128 * 4608 * 2;
    _Float16* wcat = (_Float16*)(ws + off); off += (size_t)384 * 128 * 2;
    _Float16* wa16 = (_Float16*)(ws + off); off += (size_t)128 * 128 * 2;
    _Float16* wb16 = (_Float16*)(ws + off); off += (size_t)128 * 128 * 2;
    float* sums = (float*)(ws + off); off += 128;
    float* invv = (float*)(ws + off); off += 128;

    const int nElem = BB * CC * HWsz;   // 33554432

    k_zero_sums<<<1, 32, 0, stream>>>(sums);
    k_block_sums<<<2048, 256, 0, stream>>>(x, sums, nElem);
    k_finalize_inv<<<1, 32, 0, stream>>>(sums, invv);
    k_cvt_weights<<<(WTOT + 255) / 256, 256, 0, stream>>>(w1, w2, w3, wm, wa, wb,
                                                          wcat, wm16, wa16, wb16);
    k_pack_x<<<(nElem + 255) / 256, 256, 0, stream>>>(x, y4c, nElem);

    dim3 gscale(NPIX / 64, 3);
    k_scale_convs<<<gscale, 256, 0, stream>>>(x, wcat, g1, b1, g2, b2, g3, b3, invv, y4c);
    k_mix<<<BB * HH * (WW / 128), 256, 0, stream>>>(y4c, wm16, gm, bm, z16);
    k_mlp_out<<<NPIX / 64, 256, 0, stream>>>(z16, wa16, wb16, ga, ba, gb, bb, x, out);
}
